// GraphAutoencoder_59760174957088
// MI455X (gfx1250) — compile-verified, matrix operand order checked by reading, not run on hardware
//
#include <hip/hip_runtime.h>
#include <hip/hip_bf16.h>
#include <math.h>

// ---------------------------------------------------------------------------
// GraphAutoencoder for MI455X (gfx1250, wave32, WMMA)
//   z   = sigmoid(x @ wenc^T + benc)                 [50000 x 500]
//   deg = segment_sum(ones, row); dis = deg^-0.5 (0 if deg==0)
//   aggr[col] += dis[row]*dis[col]*conv_w * z[row]   (scatter over 1.6M edges)
//   out = (aggr + z) @ wdec^T + bdec                 [50000 x 1024]
// GEMMs: v_wmma_f32_16x16x32_bf16, one wave = 16x64 strip (4 N-tiles,
// A fragment reused 4x per K-step to cut L2/HBM traffic).
// ---------------------------------------------------------------------------

#define N_NODES 50000
#define IN_DIM  1024
#define EMB     500
#define EMB_P   512      // EMB padded to multiple of 16/32

typedef __attribute__((ext_vector_type(16))) __bf16 v16bf;
typedef __attribute__((ext_vector_type(8)))  float  v8f;
typedef __attribute__((ext_vector_type(4)))  unsigned int v4u;

union Frag16 { v16bf bf; v4u u[2]; };

// ---------------------------- utility kernels ------------------------------

__global__ void zero_f32(float* __restrict__ p, size_t n) {
    size_t i = (size_t)blockIdx.x * blockDim.x + threadIdx.x;
    size_t stride = (size_t)gridDim.x * blockDim.x;
    for (; i < n; i += stride) p[i] = 0.0f;
}

__global__ void cvt_f32_bf16(const float* __restrict__ in, __bf16* __restrict__ out, size_t n) {
    size_t i = (size_t)blockIdx.x * blockDim.x + threadIdx.x;
    size_t stride = (size_t)gridDim.x * blockDim.x;
    for (; i < n; i += stride) out[i] = (__bf16)in[i];
}

// wenc [500][1024] fp32 -> [512][1024] bf16, rows >= 500 zeroed
__global__ void pack_wenc(const float* __restrict__ w, __bf16* __restrict__ out) {
    size_t i = (size_t)blockIdx.x * blockDim.x + threadIdx.x;
    size_t total = (size_t)EMB_P * IN_DIM;
    size_t stride = (size_t)gridDim.x * blockDim.x;
    for (; i < total; i += stride) {
        int n = (int)(i >> 10);            // / 1024
        int k = (int)(i & 1023);
        out[i] = (n < EMB) ? (__bf16)w[(size_t)n * IN_DIM + k] : (__bf16)0.0f;
    }
}

// wdec [1024][500] fp32 -> [1024][512] bf16, cols >= 500 zeroed
__global__ void pack_wdec(const float* __restrict__ w, __bf16* __restrict__ out) {
    size_t i = (size_t)blockIdx.x * blockDim.x + threadIdx.x;
    size_t total = (size_t)IN_DIM * EMB_P;
    size_t stride = (size_t)gridDim.x * blockDim.x;
    for (; i < total; i += stride) {
        int d = (int)(i >> 9);             // / 512
        int e = (int)(i & 511);
        out[i] = (e < EMB) ? (__bf16)w[(size_t)d * EMB + e] : (__bf16)0.0f;
    }
}

// ---------------------------- encoder GEMM ---------------------------------
// One wave computes a 16x64 strip (4 WMMA tiles) of z = sigmoid(x@wenc^T+benc).
// A = xb [50000][1024] bf16 row-major, B = wencb [512][1024] bf16 (B[k][n]=wencb[n][k]).

__global__ void enc_gemm(const __bf16* __restrict__ xb, const __bf16* __restrict__ wb,
                         const float* __restrict__ benc, float* __restrict__ z) {
    const int lane  = threadIdx.x & 31;
    const int wave  = (int)((blockIdx.x * blockDim.x + threadIdx.x) >> 5);
    const int NG    = EMB_P / 64;                 // 8 strip groups
    const int mtile = wave / NG;
    const int ng    = wave % NG;
    if (mtile >= N_NODES / 16) return;

    const int mr = lane & 15;                     // A-row (M) / B-row (N) / C-col (N)
    const int hi = lane >> 4;

    const __bf16* Ap = xb + (size_t)(mtile * 16 + mr) * IN_DIM + hi * 8;
    const __bf16* Bp[4];
#pragma unroll
    for (int j = 0; j < 4; ++j)
        Bp[j] = wb + (size_t)(ng * 64 + j * 16 + mr) * IN_DIM + hi * 8;

    const v8f vzero = {0.f, 0.f, 0.f, 0.f, 0.f, 0.f, 0.f, 0.f};
    v8f acc[4] = {vzero, vzero, vzero, vzero};

    for (int kk = 0; kk < IN_DIM; kk += 32) {
        Frag16 a;
        a.u[0] = *(const v4u*)(Ap + kk);
        a.u[1] = *(const v4u*)(Ap + kk + 16);
#pragma unroll
        for (int j = 0; j < 4; ++j) {
            Frag16 b;
            b.u[0] = *(const v4u*)(Bp[j] + kk);
            b.u[1] = *(const v4u*)(Bp[j] + kk + 16);
            acc[j] = __builtin_amdgcn_wmma_f32_16x16x32_bf16(
                         false, a.bf, false, b.bf, (short)0, acc[j], false, false);
        }
    }

#pragma unroll
    for (int j = 0; j < 4; ++j) {
        const int n = ng * 64 + j * 16 + mr;      // output feature
        const float bias = (n < EMB) ? benc[n] : 0.0f;
        float* zrow = z + (size_t)(mtile * 16 + hi * 8) * EMB_P + n;
#pragma unroll
        for (int r = 0; r < 8; ++r) {
            float v = acc[j][r] + bias;
            v = (n < EMB) ? (1.0f / (1.0f + __expf(-v))) : 0.0f;  // sigmoid, pad cols = 0
            zrow[(size_t)r * EMB_P] = v;
        }
    }
}

// ---------------------------- graph part -----------------------------------

__global__ void degree_count(const int* __restrict__ ei, unsigned int* __restrict__ deg, int E) {
    int e = blockIdx.x * blockDim.x + threadIdx.x;
    if (e < E) atomicAdd(&deg[ei[e]], 1u);        // row = ei[0][e]
}

__global__ void make_dis(const unsigned int* __restrict__ deg, float* __restrict__ dis, int n) {
    int i = blockIdx.x * blockDim.x + threadIdx.x;
    if (i < n) {
        unsigned int d = deg[i];
        dis[i] = (d > 0u) ? rsqrtf((float)d) : 0.0f;
    }
}

// One wave per edge: aggr[col] += dis[row]*dis[col]*conv_w * z[row]
__global__ void edge_scatter(const int* __restrict__ ei, const float* __restrict__ dis,
                             const float* __restrict__ z, float* __restrict__ aggr,
                             const float* __restrict__ convw, int E) {
    const int wave = (int)((blockIdx.x * blockDim.x + threadIdx.x) >> 5);
    const int lane = threadIdx.x & 31;
    if (wave >= E) return;
    const int r = ei[wave];
    const int c = ei[E + wave];
    const float nrm = dis[r] * dis[c] * convw[0];
    const float* zr = z + (size_t)r * EMB_P;
    float* ar = aggr + (size_t)c * EMB_P;
    for (int f = lane; f < EMB; f += 32) {
        unsafeAtomicAdd(ar + f, nrm * zr[f]);     // native global_atomic_add_f32
    }
}

// z2bf = bf16(aggr + z), padded cols stay exactly 0
__global__ void residual_bf16(const float* __restrict__ aggr, const float* __restrict__ z,
                              __bf16* __restrict__ out, size_t n) {
    size_t i = (size_t)blockIdx.x * blockDim.x + threadIdx.x;
    size_t stride = (size_t)gridDim.x * blockDim.x;
    for (; i < n; i += stride) out[i] = (__bf16)(aggr[i] + z[i]);
}

// ---------------------------- decoder GEMM ---------------------------------
// out[50000][1024] = z2 @ wdec^T + bdec; one wave = 16x64 strip.
// A = z2b [50000][512], B = wdecb [1024][512].

__global__ void dec_gemm(const __bf16* __restrict__ z2b, const __bf16* __restrict__ wb,
                         const float* __restrict__ bdec, float* __restrict__ out) {
    const int lane  = threadIdx.x & 31;
    const int wave  = (int)((blockIdx.x * blockDim.x + threadIdx.x) >> 5);
    const int NG    = IN_DIM / 64;                // 16 strip groups
    const int mtile = wave / NG;
    const int ng    = wave % NG;
    if (mtile >= N_NODES / 16) return;

    const int mr = lane & 15;
    const int hi = lane >> 4;

    const __bf16* Ap = z2b + (size_t)(mtile * 16 + mr) * EMB_P + hi * 8;
    const __bf16* Bp[4];
#pragma unroll
    for (int j = 0; j < 4; ++j)
        Bp[j] = wb + (size_t)(ng * 64 + j * 16 + mr) * EMB_P + hi * 8;

    const v8f vzero = {0.f, 0.f, 0.f, 0.f, 0.f, 0.f, 0.f, 0.f};
    v8f acc[4] = {vzero, vzero, vzero, vzero};

    for (int kk = 0; kk < EMB_P; kk += 32) {
        Frag16 a;
        a.u[0] = *(const v4u*)(Ap + kk);
        a.u[1] = *(const v4u*)(Ap + kk + 16);
#pragma unroll
        for (int j = 0; j < 4; ++j) {
            Frag16 b;
            b.u[0] = *(const v4u*)(Bp[j] + kk);
            b.u[1] = *(const v4u*)(Bp[j] + kk + 16);
            acc[j] = __builtin_amdgcn_wmma_f32_16x16x32_bf16(
                         false, a.bf, false, b.bf, (short)0, acc[j], false, false);
        }
    }

#pragma unroll
    for (int j = 0; j < 4; ++j) {
        const int d = ng * 64 + j * 16 + mr;
        const float bias = bdec[d];
        float* orow = out + (size_t)(mtile * 16 + hi * 8) * IN_DIM + d;
#pragma unroll
        for (int r = 0; r < 8; ++r) orow[(size_t)r * IN_DIM] = acc[j][r] + bias;
    }
}

// ---------------------------- launcher -------------------------------------

extern "C" void kernel_launch(void* const* d_in, const int* in_sizes, int n_in,
                              void* d_out, int out_size, void* d_ws, size_t ws_size,
                              hipStream_t stream) {
    const float* x      = (const float*)d_in[0];
    const int*   ei     = (const int*)d_in[1];     // [2][E]
    const float* wenc   = (const float*)d_in[2];
    const float* benc   = (const float*)d_in[3];
    const float* wdec   = (const float*)d_in[4];
    const float* bdec   = (const float*)d_in[5];
    const float* convw  = (const float*)d_in[6];
    float* outp = (float*)d_out;
    const int E = in_sizes[1] / 2;

    // ---- workspace layout (bytes, 256-aligned) ----
    char* ws = (char*)d_ws;
    size_t off = 0;
    auto alloc = [&](size_t bytes) { char* p = ws + off; off += (bytes + 255) & ~(size_t)255; return p; };
    __bf16* xb    = (__bf16*)alloc((size_t)N_NODES * IN_DIM * 2);   // 102.4 MB (reused for z2b)
    __bf16* wencb = (__bf16*)alloc((size_t)EMB_P * IN_DIM * 2);     //   1.0 MB
    __bf16* wdecb = (__bf16*)alloc((size_t)IN_DIM * EMB_P * 2);     //   1.0 MB
    float*  z     = (float*)alloc((size_t)N_NODES * EMB_P * 4);     // 102.4 MB
    float*  aggr  = (float*)alloc((size_t)N_NODES * EMB_P * 4);     // 102.4 MB
    unsigned int* deg = (unsigned int*)alloc((size_t)N_NODES * 4);
    float*  dis   = (float*)alloc((size_t)N_NODES * 4);
    __bf16* z2b   = xb;   // xb is dead after enc_gemm; reuse for z2 bf16

    const int T = 256;

    // 1) zero aggr + deg
    zero_f32<<<2048, T, 0, stream>>>(aggr, (size_t)N_NODES * EMB_P);
    zero_f32<<<64,   T, 0, stream>>>((float*)deg, (size_t)N_NODES);

    // 2) bf16 conversions / packing
    cvt_f32_bf16<<<4096, T, 0, stream>>>(x, xb, (size_t)N_NODES * IN_DIM);
    pack_wenc<<<512, T, 0, stream>>>(wenc, wencb);
    pack_wdec<<<512, T, 0, stream>>>(wdec, wdecb);

    // 3) encoder GEMM: 3125 M-tiles x 8 strips = 25000 waves, 8 waves/block
    enc_gemm<<<(N_NODES / 16) * (EMB_P / 64) / 8, T, 0, stream>>>(xb, wencb, benc, z);

    // 4) degrees + dis
    degree_count<<<(E + T - 1) / T, T, 0, stream>>>(ei, deg, E);
    make_dis<<<(N_NODES + T - 1) / T, T, 0, stream>>>(deg, dis, N_NODES);

    // 5) edge scatter (one wave per edge)
    {
        long long threads = (long long)E * 32;
        int blocks = (int)((threads + T - 1) / T);       // 200000
        edge_scatter<<<blocks, T, 0, stream>>>(ei, dis, z, aggr, convw, E);
    }

    // 6) residual + bf16
    residual_bf16<<<4096, T, 0, stream>>>(aggr, z, z2b, (size_t)N_NODES * EMB_P);

    // 7) decoder GEMM: 3125 M-tiles x 16 strips = 50000 waves
    dec_gemm<<<(N_NODES / 16) * (IN_DIM / 64) / 8, T, 0, stream>>>(z2b, wdecb, bdec, outp);
}
